// Embedder_66932770341523
// MI455X (gfx1250) — matrix operand order repaired and checked by reading, not compiled, not run on hardware
//
#include <hip/hip_runtime.h>
#include <hip/hip_bf16.h>

typedef __attribute__((ext_vector_type(16))) _Float16 v16h;
typedef __attribute__((ext_vector_type(8)))  float    v8f;

#define T_SEQ 2048
#define BATCH 64
#define HID   128
#define G3    384   // 3*HID

static __device__ __forceinline__ v8f wmma_f16(v16h a, v16h b, v8f c) {
  return __builtin_amdgcn_wmma_f32_16x16x32_f16(false, a, false, b, (short)0, c,
                                                false, false);
}

// A fragment from an f32 row (row-major, K contiguous), converting to f16.
// lane kh = lane>>4 ; K set = kbase + {0..7,16..23} + 8*kh
static __device__ __forceinline__ v16h load_a_f32(const float* __restrict__ row,
                                                  int kbase, int kh) {
  const float4 p0 = *(const float4*)(row + kbase + kh * 8);
  const float4 p1 = *(const float4*)(row + kbase + kh * 8 + 4);
  const float4 p2 = *(const float4*)(row + kbase + 16 + kh * 8);
  const float4 p3 = *(const float4*)(row + kbase + 16 + kh * 8 + 4);
  v16h a;
  a[0]=(_Float16)p0.x;  a[1]=(_Float16)p0.y;  a[2]=(_Float16)p0.z;  a[3]=(_Float16)p0.w;
  a[4]=(_Float16)p1.x;  a[5]=(_Float16)p1.y;  a[6]=(_Float16)p1.z;  a[7]=(_Float16)p1.w;
  a[8]=(_Float16)p2.x;  a[9]=(_Float16)p2.y;  a[10]=(_Float16)p2.z; a[11]=(_Float16)p2.w;
  a[12]=(_Float16)p3.x; a[13]=(_Float16)p3.y; a[14]=(_Float16)p3.z; a[15]=(_Float16)p3.w;
  return a;
}

// A fragment from an f16 row in LDS (row-major, K contiguous).
static __device__ __forceinline__ v16h load_a_f16(const _Float16* __restrict__ row,
                                                  int kbase, int kh) {
  union { uint4 u[2]; v16h h; } t;
  t.u[0] = *(const uint4*)(row + kbase + kh * 8);
  t.u[1] = *(const uint4*)(row + kbase + 16 + kh * 8);
  return t.h;
}

// B fragment: weights transposed [N][K]; p = &W[n*128 + kbase + kh*8]; K set = {0..7,16..23}.
static __device__ __forceinline__ v16h load_b_f16(const _Float16* __restrict__ p) {
  union { uint4 u[2]; v16h h; } t;
  t.u[0] = *(const uint4*)(p);
  t.u[1] = *(const uint4*)(p + 16);
  return t.h;
}

// Hardware-native sigmoid / tanh: v_exp_f32 + v_rcp_f32 (TRANS ops co-execute
// with the matrix pipe; avoids precise-division expansion and ocml tanh).
static __device__ __forceinline__ float sigm(float x) {
  return __builtin_amdgcn_rcpf(1.0f + __expf(-x));
}
static __device__ __forceinline__ float tanh_fast(float x) {
  // 1 - 2/(e^{2x}+1); exact saturation at +/-inf.
  return 1.0f - 2.0f * __builtin_amdgcn_rcpf(__expf(2.0f * x) + 1.0f);
}

// ---------------------------------------------------------------------------
// Convert weights to f16 transposed [N][K]; zero the pipeline flags.
__global__ void convert_weights(const float* __restrict__ kin,   // [128,384]
                                const float* __restrict__ rin,   // [128,384]
                                const float* __restrict__ wout,  // [128,128]
                                _Float16* __restrict__ k16T,     // [384][128]
                                _Float16* __restrict__ r16T,     // [384][128]
                                _Float16* __restrict__ o16T,     // [128][128]
                                int* __restrict__ flags) {       // [3][T]
  int i = blockIdx.x * blockDim.x + threadIdx.x;
  if (i < G3 * HID) {
    int n = i / HID, k = i - n * HID;
    k16T[i] = (_Float16)kin[(size_t)k * G3 + n];
    r16T[i] = (_Float16)rin[(size_t)k * G3 + n];
  }
  if (i < HID * HID) {
    int n = i >> 7, k = i & 127;
    o16T[i] = (_Float16)wout[(size_t)k * HID + n];
  }
  if (i < 3 * T_SEQ) flags[i] = 0;
}

// ---------------------------------------------------------------------------
// Fused, layer-pipelined GRU scan. Grid = 3 persistent workgroups (one per
// layer), 16 waves each. Wave w owns row-tile rt=w&3 and two 16-col slices
// j0 = (w>>2)*16 and j0+64. For each slice it computes ALL FOUR gate tiles
//   z  : x@Wk[:,j]     + h@Wr[:,j]      (shared accumulator)
//   r  : x@Wk[:,128+j] + h@Wr[:,128+j]  (shared accumulator)
//   xh : x@Wk[:,256+j]
//   hh : h@Wr[:,256+j]
// whose C-fragments are element-aligned, so the whole GRU gate update runs in
// registers: no staging buffer, h carried in VGPRs, one barrier per step.
__global__ __launch_bounds__(512) void gru_pipe(
    const float* __restrict__ X,             // [B,T,H]
    float* __restrict__ seq0, float* __restrict__ seq1,
    float* __restrict__ seq2,                // [T,B,H] each
    const _Float16* __restrict__ k16T,       // [384][128]
    const _Float16* __restrict__ r16T,       // [384][128]
    const float* __restrict__ bin, const float* __restrict__ brc,
    int* __restrict__ flags) {               // [3][T]
  __shared__ __align__(16) _Float16 sWk[G3 * HID];        //  96 KB
  __shared__ __align__(16) _Float16 sWr[G3 * HID];        //  96 KB
  __shared__ __align__(16) _Float16 sH[2][BATCH * HID];   //  32 KB (dbl-buf)

  const int layer = blockIdx.x;              // 0..2
  const int tid = threadIdx.x;
  const int lane = tid & 31;
  const int wave = tid >> 5;                 // 0..15
  const int l15 = lane & 15, kh = lane >> 4;
  const int rt = wave & 3;                   // row tile 0..3
  const int jg = (wave >> 2) & 3;            // col-slice group 0..3

  float* seq_out = (layer == 0) ? seq0 : (layer == 1) ? seq1 : seq2;
  const float* seq_in = (layer == 1) ? seq0 : seq1;  // unused for layer 0
  int* f_self = flags + layer * T_SEQ;
  int* f_prev = flags + (layer - 1) * T_SEQ;

  // Stage both weight matrices; zero both h buffers.
  for (int i = tid; i < (G3 * HID) / 8; i += 512) {
    ((uint4*)sWk)[i] = ((const uint4*)k16T)[i];
    ((uint4*)sWr)[i] = ((const uint4*)r16T)[i];
  }
  for (int i = tid; i < BATCH * HID; i += 512) {
    sH[0][i] = (_Float16)0.0f;
    sH[1][i] = (_Float16)0.0f;
  }
  __syncthreads();

  const int b_row = rt * 16 + l15;           // batch row for A fragments

  // Per-job column and fused biases (loop-invariant, registers).
  int   ncol[2];
  float bZ[2], bR[2], bXH[2], bHH[2];
#pragma unroll
  for (int jj = 0; jj < 2; ++jj) {
    ncol[jj] = (jg + 4 * jj) * 16 + l15;     // j0 + l15, j0+64 + l15
    bZ[jj]  = bin[ncol[jj]] + brc[ncol[jj]];
    bR[jj]  = bin[HID + ncol[jj]] + brc[HID + ncol[jj]];
    bXH[jj] = bin[2 * HID + ncol[jj]];
    bHH[jj] = brc[2 * HID + ncol[jj]];
  }

  // Hidden state owned by this lane: rows rt*16 + r + 8*kh, col ncol[jj].
  float hreg[2][8];
#pragma unroll
  for (int jj = 0; jj < 2; ++jj)
#pragma unroll
    for (int r = 0; r < 8; ++r) hreg[jj][r] = 0.0f;

  for (int t = 0; t < T_SEQ; ++t) {
    if (layer > 0) {                         // wait for producer's h_t
      if (tid == 0) {
        while (__hip_atomic_load(&f_prev[t], __ATOMIC_ACQUIRE,
                                 __HIP_MEMORY_SCOPE_AGENT) == 0)
          __builtin_amdgcn_s_sleep(1);
      }
      __syncthreads();
    }

    const float* xRow = (layer == 0)
        ? X + ((size_t)b_row * T_SEQ + t) * HID
        : seq_in + ((size_t)t * BATCH + b_row) * HID;
    const _Float16* hCur = &sH[t & 1][0];
    _Float16* hNxt = &sH[(t + 1) & 1][0];

    v8f aZ[2], aR[2], aXH[2], aHH[2];
#pragma unroll
    for (int jj = 0; jj < 2; ++jj) {
      aZ[jj] = (v8f){0,0,0,0,0,0,0,0};  aR[jj]  = (v8f){0,0,0,0,0,0,0,0};
      aXH[jj] = (v8f){0,0,0,0,0,0,0,0}; aHH[jj] = (v8f){0,0,0,0,0,0,0,0};
    }

#pragma unroll
    for (int ks = 0; ks < 4; ++ks) {
      const int ko = ks * 32 + kh * 8;
      v16h ax = load_a_f32(xRow, ks * 32, kh);
      v16h ah = load_a_f16(hCur + (size_t)b_row * HID, ks * 32, kh);
#pragma unroll
      for (int jj = 0; jj < 2; ++jj) {
        const int n0 = ncol[jj];
        aZ[jj]  = wmma_f16(ax, load_b_f16(sWk + (size_t)n0 * HID + ko), aZ[jj]);
        aZ[jj]  = wmma_f16(ah, load_b_f16(sWr + (size_t)n0 * HID + ko), aZ[jj]);
        aR[jj]  = wmma_f16(ax, load_b_f16(sWk + (size_t)(HID + n0) * HID + ko), aR[jj]);
        aR[jj]  = wmma_f16(ah, load_b_f16(sWr + (size_t)(HID + n0) * HID + ko), aR[jj]);
        aXH[jj] = wmma_f16(ax, load_b_f16(sWk + (size_t)(2 * HID + n0) * HID + ko), aXH[jj]);
        aHH[jj] = wmma_f16(ah, load_b_f16(sWr + (size_t)(2 * HID + n0) * HID + ko), aHH[jj]);
      }
    }

    // Gate update entirely in registers; publish h_t.
    float* ho = seq_out + (size_t)t * BATCH * HID;
#pragma unroll
    for (int jj = 0; jj < 2; ++jj) {
      const int col = ncol[jj];
#pragma unroll
      for (int r = 0; r < 8; ++r) {
        const int row = rt * 16 + r + 8 * kh;          // batch index
        float z    = sigm(aZ[jj][r] + bZ[jj]);
        float rr   = sigm(aR[jj][r] + bR[jj]);
        float cand = tanh_fast(aXH[jj][r] + bXH[jj] + rr * (aHH[jj][r] + bHH[jj]));
        float hn = z * hreg[jj][r] + (1.0f - z) * cand;
        hreg[jj][r] = hn;
        hNxt[row * HID + col] = (_Float16)hn;
        ho[row * HID + col] = hn;
      }
    }
    if (layer == 0 && t + 1 < T_SEQ)         // warm caches for next X rows
      __builtin_prefetch(X + ((size_t)b_row * T_SEQ + t + 1) * HID + lane * 4, 0, 1);

    if (layer < 2) {
      __threadfence();                        // make h_t device-visible
      __syncthreads();
      if (tid == 0)
        __hip_atomic_store(&f_self[t], 1, __ATOMIC_RELEASE,
                           __HIP_MEMORY_SCOPE_AGENT);
    } else {
      __syncthreads();                        // protect sH double buffer only
    }
  }
}

// ---------------------------------------------------------------------------
// Output projection: out[b][t][:] = sigmoid(seq2[t][b][:] @ W_out + b_out)
__global__ __launch_bounds__(256) void out_proj(
    const float* __restrict__ seq,           // [T][64][128]
    const _Float16* __restrict__ o16T,       // [128][128] transposed
    const float* __restrict__ bout, float* __restrict__ out) {
  const int t = blockIdx.x;
  const int lane = threadIdx.x & 31;
  const int wave = threadIdx.x >> 5;
  const int rt = wave >> 1;
  const int ct0 = (wave & 1) * 4;
  const int l15 = lane & 15, kh = lane >> 4;

  const int b = rt * 16 + l15;
  const float* Arow = seq + ((size_t)t * BATCH + b) * HID;

  v8f acc[4];
#pragma unroll
  for (int c = 0; c < 4; ++c) acc[c] = (v8f){0,0,0,0,0,0,0,0};
#pragma unroll
  for (int ks = 0; ks < 4; ++ks) {
    v16h a = load_a_f32(Arow, ks * 32, kh);
#pragma unroll
    for (int c = 0; c < 4; ++c) {
      int n = (ct0 + c) * 16 + l15;
      v16h bf = load_b_f16(o16T + (size_t)n * HID + ks * 32 + kh * 8);
      acc[c] = wmma_f16(a, bf, acc[c]);
    }
  }
#pragma unroll
  for (int c = 0; c < 4; ++c) {
    int col = (ct0 + c) * 16 + l15;
    float bb = bout[col];
#pragma unroll
    for (int r = 0; r < 8; ++r) {
      int row = rt * 16 + r + 8 * kh;
      out[((size_t)row * T_SEQ + t) * HID + col] = sigm(acc[c][r] + bb);
    }
  }
}

// ---------------------------------------------------------------------------
extern "C" void kernel_launch(void* const* d_in, const int* in_sizes, int n_in,
                              void* d_out, int out_size, void* d_ws, size_t ws_size,
                              hipStream_t stream) {
  const float* X        = (const float*)d_in[0];  // [64, 2048, 128]
  const float* kernel   = (const float*)d_in[1];  // [128, 384]
  const float* rec_k    = (const float*)d_in[2];  // [128, 384]
  const float* bias_in  = (const float*)d_in[3];  // [384]
  const float* bias_rec = (const float*)d_in[4];  // [384]
  const float* W_out    = (const float*)d_in[5];  // [128, 128]
  const float* b_out    = (const float*)d_in[6];  // [128]
  float* out = (float*)d_out;                     // [64, 2048, 128]

  char* ws = (char*)d_ws;
  size_t off = 0;
  _Float16* k16T = (_Float16*)(ws + off); off += (size_t)G3 * HID * 2;
  _Float16* r16T = (_Float16*)(ws + off); off += (size_t)G3 * HID * 2;
  _Float16* o16T = (_Float16*)(ws + off); off += (size_t)HID * HID * 2;
  int* flags = (int*)(ws + off);          off += (size_t)3 * T_SEQ * 4;
  off = (off + 255) & ~(size_t)255;
  float* seq0 = (float*)(ws + off); off += (size_t)T_SEQ * BATCH * HID * 4;
  float* seq1 = (float*)(ws + off); off += (size_t)T_SEQ * BATCH * HID * 4;
  float* seq2 = (float*)(ws + off); off += (size_t)T_SEQ * BATCH * HID * 4;

  convert_weights<<<(G3 * HID + 255) / 256, 256, 0, stream>>>(
      kernel, rec_k, W_out, k16T, r16T, o16T, flags);

  gru_pipe<<<3, 512, 0, stream>>>(X, seq0, seq1, seq2, k16T, r16T,
                                  bias_in, bias_rec, flags);

  out_proj<<<T_SEQ, 256, 0, stream>>>(seq2, o16T, b_out, out);
}
